// DeformAtten2D_88184268522200
// MI455X (gfx1250) — matrix operand order repaired and checked by reading, not compiled, not run on hardware
//
#include <hip/hip_runtime.h>
#include <stdint.h>

typedef __attribute__((ext_vector_type(16))) __bf16 v16bf;
typedef __attribute__((ext_vector_type(8)))  float  v8f;
typedef __attribute__((ext_vector_type(4)))  unsigned int u32x4;
typedef __attribute__((ext_vector_type(2)))  unsigned int u32x2;
typedef __attribute__((ext_vector_type(4)))  int i32x4;
typedef __attribute__((ext_vector_type(4)))  float f32x4;
typedef unsigned short u16b;

static constexpr int   HW    = 128 * 128;            // 16384
static constexpr float SCALE = 0.08838834764831845f; // 128^-0.5

// ---- gfx1250 async global->LDS path (ASYNCcnt), with safe fallback --------
#if defined(__has_builtin)
# if __has_builtin(__builtin_amdgcn_global_load_async_to_lds_b128)
#  define HAVE_ASYNC_LDS 1
# endif
#endif
#ifndef HAVE_ASYNC_LDS
# define HAVE_ASYNC_LDS 0
#endif

typedef __attribute__((address_space(1))) i32x4* g1p_t;   // global 128b chunk
typedef __attribute__((address_space(3))) i32x4* l3p_t;   // LDS 128b chunk

static __device__ __forceinline__ void async_cp16(const void* gsrc, void* ldst) {
#if HAVE_ASYNC_LDS
  __builtin_amdgcn_global_load_async_to_lds_b128((g1p_t)gsrc, (l3p_t)ldst, 0, 0);
#else
  *(u32x4*)ldst = *(const u32x4*)gsrc;
#endif
}
static __device__ __forceinline__ void async_fence() {
#if HAVE_ASYNC_LDS
# if __has_builtin(__builtin_amdgcn_s_wait_asynccnt)
  __builtin_amdgcn_s_wait_asynccnt(0);
# else
  asm volatile("s_wait_asynccnt 0x0" ::: "memory");
# endif
#endif
}

static __device__ __forceinline__ u16b f2bf(float f) {
  union { float f; uint32_t u; } v; v.f = f;
  uint32_t u = v.u;
  u += 0x7FFFu + ((u >> 16) & 1u);   // round-to-nearest-even
  return (u16b)(u >> 16);
}
static __device__ __forceinline__ float bf2f(u16b h) {
  union { uint32_t u; float f; } v; v.u = ((uint32_t)h) << 16;
  return v.f;
}
static __device__ __forceinline__ unsigned pack2(float a, float b) {
  return (unsigned)f2bf(a) | ((unsigned)f2bf(b) << 16);
}

union Frag16 { v16bf v; u32x4 q[2]; };

// ---------------------------------------------------------------------------
// Weight prep: bf16 copies of Wq/Wk/Wv/Wo; fold 1x1 conv (Wc2) into 5x5 conv
// (Wc1): Wfold[d][t][c] = sum_o Wc2[d,o]*Wc1[o,c,t];  bfold[d] = Wc2[d,:]·bc1
// ---------------------------------------------------------------------------
__global__ __launch_bounds__(256) void prep_weights(
    const float* __restrict__ Wq, const float* __restrict__ Wk,
    const float* __restrict__ Wv, const float* __restrict__ Wo,
    const float* __restrict__ Wc1, const float* __restrict__ bc1,
    const float* __restrict__ Wc2,
    u16b* __restrict__ Wq_bf, u16b* __restrict__ Wk_bf,
    u16b* __restrict__ Wv_bf, u16b* __restrict__ Wo_bf,
    float* __restrict__ Wfold, float* __restrict__ bfold)
{
  const int tid = threadIdx.x;
  for (int i = tid; i < 16384; i += 256) {
    Wq_bf[i] = f2bf(Wq[i]);
    Wk_bf[i] = f2bf(Wk[i]);
    Wv_bf[i] = f2bf(Wv[i]);
    Wo_bf[i] = f2bf(Wo[i]);
  }
  for (int i = tid; i < 6400; i += 256) {        // i = d*3200 + t*128 + c
    const int d = i / 3200;
    const int r = i % 3200;
    const int t = r / 128;
    const int c = r % 128;
    float acc = 0.f;
    for (int o = 0; o < 128; ++o)
      acc += Wc2[d * 128 + o] * Wc1[((size_t)(o * 128 + c)) * 25 + t];
    Wfold[i] = acc;
  }
  if (tid < 2) {
    float acc = 0.f;
    for (int o = 0; o < 128; ++o) acc += Wc2[tid * 128 + o] * bc1[o];
    bfold[tid] = acc;
  }
}

// ---------------------------------------------------------------------------
// Generic K=128 GEMM:  D[m,n] = A[m,:]·W[n,:] + bias[n].
// The 128x128 bf16 weight is async-staged once per workgroup into LDS;
// B fragments come from ds_load_b128.  Wave = one 16-row M tile x 8 N tiles.
// MODE 0: A is fp32 (x), converted to bf16 fragments in registers;
//         store position-major bf16 (out_pm) AND head-major bf16 (out_h)  [q]
// MODE 1: store head-major bf16 (out_h)                                   [k]
// MODE 2: add rel_bias[n,h], store transposed head-major (head,d=w,j=h)   [v]
// MODE 3: store fp32 position-major (outf)                            [final]
// ---------------------------------------------------------------------------
template <int MODE>
__global__ __launch_bounds__(256) void gemm128(
    const u16b* __restrict__ A, const float* __restrict__ Af,
    const u16b* __restrict__ Wb,
    const float* __restrict__ bias, const float* __restrict__ relb,
    u16b* __restrict__ out_pm, u16b* __restrict__ out_h,
    float* __restrict__ outf)
{
  constexpr bool AF32 = (MODE == 0);
  __shared__ u16b s_w[16384];                       // 32 KB weight tile

  for (int i = threadIdx.x; i < 2048; i += 256)
    async_cp16((const u32x4*)Wb + i, (u32x4*)s_w + i);
  async_fence();
  __syncthreads();

  const int lane = threadIdx.x & 31;
  const int wave = threadIdx.x >> 5;
  const int m0   = (blockIdx.x * 8 + wave) * 16;
  const int n    = lane & 15;   // B/D column, and A row within tile
  const int s    = lane >> 4;   // K-half / M-half selector

  v8f z = {};
  v8f acc[8];
#pragma unroll
  for (int t = 0; t < 8; ++t) acc[t] = z;

  const u16b*  arow = AF32 ? nullptr : (A + (size_t)(m0 + n) * 128);
  const float* arf  = AF32 ? (Af + (size_t)(m0 + n) * 128) : nullptr;

#pragma unroll
  for (int kc = 0; kc < 4; ++kc) {
    Frag16 fa;  // A 16x32 bf16 layout: lane holds K [s*8,+8) and [16+s*8,+8)
    if (AF32) {
      const f32x4 a0 = *(const f32x4*)(arf + kc * 32 +      s * 8);
      const f32x4 a1 = *(const f32x4*)(arf + kc * 32 +      s * 8 + 4);
      const f32x4 a2 = *(const f32x4*)(arf + kc * 32 + 16 + s * 8);
      const f32x4 a3 = *(const f32x4*)(arf + kc * 32 + 16 + s * 8 + 4);
      u32x4 p0, p1;
      p0.x = pack2(a0.x, a0.y); p0.y = pack2(a0.z, a0.w);
      p0.z = pack2(a1.x, a1.y); p0.w = pack2(a1.z, a1.w);
      p1.x = pack2(a2.x, a2.y); p1.y = pack2(a2.z, a2.w);
      p1.z = pack2(a3.x, a3.y); p1.w = pack2(a3.z, a3.w);
      fa.q[0] = p0; fa.q[1] = p1;
    } else {
      fa.q[0] = *(const u32x4*)(arow + kc * 32 +      s * 8);
      fa.q[1] = *(const u32x4*)(arow + kc * 32 + 16 + s * 8);
    }
#pragma unroll
    for (int t = 0; t < 8; ++t) {
      const u16b* brow = s_w + (size_t)(t * 16 + n) * 128 + kc * 32;
      Frag16 fb;  // B 32x16: lane n holds K [s*8,+8) and [16+s*8,+8)
      fb.q[0] = *(const u32x4*)(brow + s * 8);
      fb.q[1] = *(const u32x4*)(brow + 16 + s * 8);
      acc[t] = __builtin_amdgcn_wmma_f32_16x16x32_bf16(
          false, fa.v, false, fb.v, (short)0, acc[t], false, false);
    }
  }

  // tile position decode (tiles never cross a row since 128 % 16 == 0)
  const int b  = m0 >> 14;
  const int rm = m0 & 16383;
  const int h  = rm >> 7;
  const int w0 = rm & 127;

#pragma unroll
  for (int t = 0; t < 8; ++t) {
    const int ng = t * 16 + n;
    float bb = bias[ng];
    if (MODE == 2) bb += relb[ng * 128 + h];   // rel_bias[(1,C,H,1)] -> [c*H+h]
    float val[8];
#pragma unroll
    for (int r = 0; r < 8; ++r) val[r] = acc[t][r] + bb;   // D row m = 8*s+r

    if (MODE == 0 || MODE == 1) {
      // head-major: [(b*C+c)*HW + h*128 + w]; lane's 8 values are contiguous w
      u32x4 pk;
      pk.x = pack2(val[0], val[1]); pk.y = pack2(val[2], val[3]);
      pk.z = pack2(val[4], val[5]); pk.w = pack2(val[6], val[7]);
      *(u32x4*)(out_h + (size_t)(b * 128 + ng) * HW + h * 128 + w0 + 8 * s) = pk;
      if (MODE == 0) {
#pragma unroll
        for (int r = 0; r < 8; ++r)
          out_pm[(size_t)(m0 + 8 * s + r) * 128 + ng] = f2bf(val[r]);
      }
    } else if (MODE == 2) {
      // v transposed: [(b*C+c)*HW + w*128 + h] so attn@v B-fragment is contiguous
#pragma unroll
      for (int r = 0; r < 8; ++r)
        out_h[(size_t)(b * 128 + ng) * HW + (w0 + 8 * s + r) * 128 + h] = f2bf(val[r]);
    } else {
#pragma unroll
      for (int r = 0; r < 8; ++r)
        outf[(size_t)(m0 + 8 * s + r) * 128 + ng] = val[r];
    }
  }
}

// ---------------------------------------------------------------------------
// Folded offset conv: 5x5 over q (bf16 position-major) with 2 output channels,
// then tanh * 5.  0.84 GFLOP total (64x less than the unfolded conv).
// ---------------------------------------------------------------------------
__global__ __launch_bounds__(256) void offset_conv(
    const u16b* __restrict__ q_pm, const float* __restrict__ Wfold,
    const float* __restrict__ bfold, float* __restrict__ offs)
{
  const int p = blockIdx.x * 256 + threadIdx.x;          // 65536 positions
  const int b = p >> 14, rm = p & 16383, h = rm >> 7, w = rm & 127;
  float a0 = bfold[0], a1 = bfold[1];
  for (int t = 0; t < 25; ++t) {
    const int hh = h + t / 5 - 2;
    const int ww = w + t % 5 - 2;
    if ((unsigned)hh < 128u && (unsigned)ww < 128u) {
      const u16b*  qr  = q_pm + ((size_t)(b * HW + hh * 128 + ww)) * 128;
      const float* w0p = Wfold + t * 128;
      const float* w1p = Wfold + 3200 + t * 128;
#pragma unroll 8
      for (int c = 0; c < 128; ++c) {
        const float qv = bf2f(qr[c]);
        a0 += qv * w0p[c];
        a1 += qv * w1p[c];
      }
    }
  }
  offs[2 * p]     = tanhf(a0) * 5.0f;
  offs[2 * p + 1] = tanhf(a1) * 5.0f;
}

// ---------------------------------------------------------------------------
// Bilinear grid-sample (zero padding, faithful to the reference's /(H-1)
// normalization).  Reads fp32 x directly.  One wave per position, lane = 4 ch.
// ---------------------------------------------------------------------------
__global__ __launch_bounds__(256) void grid_sample_k(
    const float* __restrict__ xf, const float* __restrict__ offs,
    u16b* __restrict__ xs)
{
  const int lane = threadIdx.x & 31;
  const int wave = threadIdx.x >> 5;
  const int p = blockIdx.x * 8 + wave;
  const int b = p >> 14, rm = p & 16383, h = rm >> 7, w = rm & 127;

  const float off0 = offs[2 * p], off1 = offs[2 * p + 1];
  const float xg = ((float)w + off0) * (128.0f / 127.0f) - 0.5f;
  const float yg = ((float)h + off1) * (128.0f / 127.0f) - 0.5f;
  const float x0 = floorf(xg), y0 = floorf(yg);
  const int c0 = lane * 4;

  float a0 = 0.f, a1 = 0.f, a2 = 0.f, a3 = 0.f;
#pragma unroll
  for (int dy = 0; dy < 2; ++dy) {
#pragma unroll
    for (int dx = 0; dx < 2; ++dx) {
      const float xi = x0 + (float)dx, yi = y0 + (float)dy;
      const float wg = (1.0f - fabsf(xg - xi)) * (1.0f - fabsf(yg - yi));
      if (xi >= 0.f && xi <= 127.f && yi >= 0.f && yi <= 127.f) {  // uniform per wave
        const int ix = (int)xi, iy = (int)yi;
        const f32x4 v = *(const f32x4*)(xf + ((size_t)(b * HW + iy * 128 + ix)) * 128 + c0);
        a0 += wg * v.x; a1 += wg * v.y; a2 += wg * v.z; a3 += wg * v.w;
      }
    }
  }
  u32x2 pk; pk.x = pack2(a0, a1); pk.y = pack2(a2, a3);
  *(u32x2*)(xs + (size_t)p * 128 + c0) = pk;
}

// ---------------------------------------------------------------------------
// Attention: one workgroup (8 waves) per head (B*C = 512 heads).
// K and V for the head are async-staged once into LDS (32 KB each) and
// shared by all 8 waves; scores = q·kT (WMMA), softmax (shfl_xor within
// 16-lane groups), attn restaged D-layout->A-layout through LDS, out = attn·v
// (WMMA, v pre-transposed).  Output stored head-major == the reference's raw
// (B*C,H,W)->(B,H,W,C) reshape, so the final GEMM reads it directly.
// ---------------------------------------------------------------------------
__global__ __launch_bounds__(256) void attn_kernel(
    const u16b* __restrict__ qh, const u16b* __restrict__ kh,
    const u16b* __restrict__ vt, u16b* __restrict__ ao)
{
  __shared__ u16b s_k[16384];     // 32 KB: K head tile (row j, contiguous w)
  __shared__ u16b s_v[16384];     // 32 KB: V head tile (row d, contiguous j)
  __shared__ u16b s_attn[16384];  // 32 KB: per-wave 16x128 attn rows

  const int lane = threadIdx.x & 31;
  const int wave = threadIdx.x >> 5;
  const int n = lane & 15;
  const int s = lane >> 4;
  const int i0 = wave * 16;
  const size_t hb = (size_t)blockIdx.x * HW;

  {
    const u32x4* srck = (const u32x4*)(kh + hb);
    const u32x4* srcv = (const u32x4*)(vt + hb);
    u32x4* dk = (u32x4*)s_k;
    u32x4* dv = (u32x4*)s_v;
    for (int i = threadIdx.x; i < 2048; i += 256) {
      async_cp16(srck + i, dk + i);
      async_cp16(srcv + i, dv + i);
    }
    async_fence();
  }
  __syncthreads();

  v8f z = {};
  v8f S[8];
#pragma unroll
  for (int t = 0; t < 8; ++t) S[t] = z;

#pragma unroll
  for (int kc = 0; kc < 4; ++kc) {
    const u16b* ar = qh + hb + (size_t)(i0 + n) * 128 + kc * 32;
    Frag16 fa;
    fa.q[0] = *(const u32x4*)(ar + s * 8);
    fa.q[1] = *(const u32x4*)(ar + 16 + s * 8);
#pragma unroll
    for (int t = 0; t < 8; ++t) {
      const u16b* br = s_k + (size_t)(t * 16 + n) * 128 + kc * 32;
      Frag16 fb;
      fb.q[0] = *(const u32x4*)(br + s * 8);
      fb.q[1] = *(const u32x4*)(br + 16 + s * 8);
      S[t] = __builtin_amdgcn_wmma_f32_16x16x32_bf16(
          false, fa.v, false, fb.v, (short)0, S[t], false, false);
    }
  }

  // softmax over j: row m=8*s+r spans the 8 tiles and the 16 lanes of group s
  float rinv[8];
#pragma unroll
  for (int r = 0; r < 8; ++r) {
    float m = -3.4e38f;
#pragma unroll
    for (int t = 0; t < 8; ++t) { S[t][r] *= SCALE; m = fmaxf(m, S[t][r]); }
#pragma unroll
    for (int msk = 1; msk < 16; msk <<= 1) m = fmaxf(m, __shfl_xor(m, msk, 32));
    float sum = 0.f;
#pragma unroll
    for (int t = 0; t < 8; ++t) { const float e = __expf(S[t][r] - m); S[t][r] = e; sum += e; }
#pragma unroll
    for (int msk = 1; msk < 16; msk <<= 1) sum += __shfl_xor(sum, msk, 32);
    rinv[r] = 1.0f / sum;
  }

  u16b* sa = s_attn + wave * 2048;
#pragma unroll
  for (int t = 0; t < 8; ++t)
#pragma unroll
    for (int r = 0; r < 8; ++r)
      sa[(8 * s + r) * 128 + t * 16 + n] = f2bf(S[t][r] * rinv[r]);
  __syncthreads();

  v8f O[8];
#pragma unroll
  for (int t = 0; t < 8; ++t) O[t] = z;

#pragma unroll
  for (int kc = 0; kc < 4; ++kc) {
    const u16b* ar = sa + n * 128 + kc * 32;
    Frag16 fa;
    fa.q[0] = *(const u32x4*)(ar + s * 8);
    fa.q[1] = *(const u32x4*)(ar + 16 + s * 8);
#pragma unroll
    for (int t = 0; t < 8; ++t) {
      const u16b* br = s_v + (size_t)(t * 16 + n) * 128 + kc * 32;
      Frag16 fb;
      fb.q[0] = *(const u32x4*)(br + s * 8);
      fb.q[1] = *(const u32x4*)(br + 16 + s * 8);
      O[t] = __builtin_amdgcn_wmma_f32_16x16x32_bf16(
          false, fa.v, false, fb.v, (short)0, O[t], false, false);
    }
  }

#pragma unroll
  for (int t = 0; t < 8; ++t) {
    const int d = t * 16 + n;
#pragma unroll
    for (int r = 0; r < 8; ++r)
      ao[hb + (size_t)(i0 + 8 * s + r) * 128 + d] = f2bf(O[t][r]);
  }
}

// ---------------------------------------------------------------------------
extern "C" void kernel_launch(void* const* d_in, const int* in_sizes, int n_in,
                              void* d_out, int out_size, void* d_ws, size_t ws_size,
                              hipStream_t stream)
{
  (void)in_sizes; (void)n_in; (void)out_size; (void)ws_size;

  const float* x    = (const float*)d_in[0];
  // d_in[1] = cycle_index (dead path in cycle_mode='None')
  const float* Wq   = (const float*)d_in[2];
  const float* bq   = (const float*)d_in[3];
  const float* Wk   = (const float*)d_in[4];
  const float* bk   = (const float*)d_in[5];
  const float* Wv   = (const float*)d_in[6];
  const float* bv   = (const float*)d_in[7];
  const float* Wo   = (const float*)d_in[8];
  const float* bo   = (const float*)d_in[9];
  const float* Wc1  = (const float*)d_in[10];
  const float* bc1  = (const float*)d_in[11];
  const float* Wc2  = (const float*)d_in[12];
  const float* relb = (const float*)d_in[13];
  float* out = (float*)d_out;

  char* ws = (char*)d_ws;
  u16b* ao   = (u16b*)(ws);                        // 16 MB attn output
  u16b* q_pm = (u16b*)(ws + (16u << 20));          // 16 MB (later aliased: xs)
  u16b* q_h  = (u16b*)(ws + (32u << 20));          // 16 MB
  u16b* k_h  = (u16b*)(ws + (48u << 20));          // 16 MB
  u16b* v_t  = (u16b*)(ws + (64u << 20));          // 16 MB
  char* wb   = ws + (80u << 20);
  u16b*  Wq_bf = (u16b*)(wb);
  u16b*  Wk_bf = (u16b*)(wb + 32768);
  u16b*  Wv_bf = (u16b*)(wb + 65536);
  u16b*  Wo_bf = (u16b*)(wb + 98304);
  float* Wfold = (float*)(wb + 131072);            // 2*25*128 fp32
  float* bfold = (float*)(wb + 131072 + 51200);    // 2 fp32
  float* offs  = (float*)(wb + 131072 + 51456);    // 65536*2 fp32
  u16b* xs = q_pm;   // x_sampled reuses q buffer (q consumed by offset_conv)

  prep_weights<<<1, 256, 0, stream>>>(Wq, Wk, Wv, Wo, Wc1, bc1, Wc2,
                                      Wq_bf, Wk_bf, Wv_bf, Wo_bf, Wfold, bfold);
  gemm128<0><<<512, 256, 0, stream>>>(nullptr, x, Wq_bf, bq, nullptr, q_pm, q_h, nullptr);
  offset_conv<<<256, 256, 0, stream>>>(q_pm, Wfold, bfold, offs);
  grid_sample_k<<<8192, 256, 0, stream>>>(x, offs, xs);
  gemm128<1><<<512, 256, 0, stream>>>(xs, nullptr, Wk_bf, bk, nullptr, nullptr, k_h, nullptr);
  gemm128<2><<<512, 256, 0, stream>>>(xs, nullptr, Wv_bf, bv, relb, nullptr, v_t, nullptr);
  attn_kernel<<<512, 256, 0, stream>>>(q_h, k_h, v_t, ao);
  gemm128<3><<<512, 256, 0, stream>>>(ao, nullptr, Wo_bf, bo, nullptr, nullptr, nullptr, out);
}
